// MilliesRNN_25572235281183
// MI455X (gfx1250) — matrix operand order repaired and checked by reading, not compiled, not run on hardware
//
#include <hip/hip_runtime.h>
#include <hip/hip_bf16.h>
#include <math.h>

// Problem constants (from reference)
#define B_  256
#define T_  512
#define L_  128
#define O_  64
#define H_  576      // 512 visible + 64 thalamic feedback
#define BT  16       // batch rows per workgroup (== WMMA M)
#define HP  592      // LDS row stride for H-wide bf16 buffers (16B-aligned rows)
#define NTHREADS 288 // 9 wave32s; 36 N-tiles -> 4 per wave

// Packed-weight layout: per 32x16 K-by-N tile, lane l owns column n = nt*16 + (l&15)
// and 16 K values k = kt*32 + 8*(l>>4) + (j<8 ? j : 8+j), stored as 32 contiguous bytes.
constexpr size_t OFS_I2H  = 0;        // Kt=4 (K=128), Nt=32 -> 65536
constexpr size_t OFS_H2H  = 65536;    // Kt=18 (K=576), Nt=36 -> 331776
constexpr size_t OFS_H2O  = 397312;   // Kt=18, Nt=4 -> 36864
constexpr size_t OFS_THAL = 434176;   // Kt=5 (K 129->160), Nt=4 -> 10240
constexpr size_t OFS_I2HD = 444416;   // Kt=3 (K 65->96), Nt=36 -> 55296
constexpr size_t OFS_H2HD = 499712;   // Kt=18, Nt=36 -> 331776
constexpr size_t OFS_H2OD = 831488;   // Kt=18, Nt=4 -> 36864
constexpr size_t WPACK_TOTAL = 868352; // ushorts (~1.7 MB) -- resident in 192MB L2

typedef __attribute__((ext_vector_type(16))) __bf16 v16bf;
typedef __attribute__((ext_vector_type(8)))  float  v8f;

union FragU {
  v16bf v;
  uint4 u[2];
  unsigned short s[16];
};

__device__ __forceinline__ unsigned short f2bf(float f) {
  unsigned int u = __float_as_uint(f);
  u += 0x7FFFu + ((u >> 16) & 1u);       // round-to-nearest-even
  return (unsigned short)(u >> 16);
}

__device__ __forceinline__ float retanh_f(float x) {
  if (x <= 0.0f) return 0.0f;
#if __has_builtin(__builtin_amdgcn_tanhf)
  return __builtin_amdgcn_tanhf(x);      // V_TANH_F32 trans op
#elif __has_builtin(__builtin_amdgcn_tanh_f32)
  return __builtin_amdgcn_tanh_f32(x);
#else
  return tanhf(x);
#endif
}

__device__ __forceinline__ v8f wmma_bf16(v16bf a, v16bf b, v8f c) {
  // (neg_a, A, neg_b, B, c_mod, C, reuse_a, reuse_b)
  return __builtin_amdgcn_wmma_f32_16x16x32_bf16(false, a, false, b, (short)0, c, false, false);
}

// B fragment: 32 contiguous bytes per lane from the packed weight tile.
__device__ __forceinline__ v16bf load_bfrag(const unsigned short* tile, int lane) {
  FragU f;
  const uint4* p = (const uint4*)(tile + lane * 16);
  f.u[0] = p[0];
  f.u[1] = p[1];
  return f.v;
}

// A fragment from LDS row-major bf16: documented interleaved layout.
// elements 0..7  <- K = kt*32 + 8*half + [0..7]
// elements 8..15 <- K = kt*32 + 16 + 8*half + [0..7]
__device__ __forceinline__ v16bf load_afrag_lds(const unsigned short* row, int kt, int half) {
  FragU f;
  f.u[0] = *(const uint4*)(row + kt * 32 + 8 * half);
  f.u[1] = *(const uint4*)(row + kt * 32 + 16 + 8 * half);
  return f.v;
}

// ---------------------------------------------------------------------------
// One-shot weight packer: fp32 (out,in) row-major -> bf16 B-fragment tiles.
// ---------------------------------------------------------------------------
extern "C" __global__ void __launch_bounds__(256)
pack_weights_kernel(const float* __restrict__ i2h_w, const float* __restrict__ h2h_w,
                    const float* __restrict__ h2o_w, const float* __restrict__ thal_w,
                    const float* __restrict__ i2hd_w, const float* __restrict__ h2hd_w,
                    const float* __restrict__ h2od_w, unsigned short* __restrict__ dst)
{
  size_t idx = (size_t)blockIdx.x * blockDim.x + threadIdx.x;
  if (idx >= WPACK_TOTAL) return;
  const float* W; int Kact, Nt; size_t base;
  if      (idx < OFS_H2H)  { W = i2h_w;  Kact = 128; Nt = 32; base = OFS_I2H;  }
  else if (idx < OFS_H2O)  { W = h2h_w;  Kact = 576; Nt = 36; base = OFS_H2H;  }
  else if (idx < OFS_THAL) { W = h2o_w;  Kact = 576; Nt = 4;  base = OFS_H2O;  }
  else if (idx < OFS_I2HD) { W = thal_w; Kact = 129; Nt = 4;  base = OFS_THAL; }
  else if (idx < OFS_H2HD) { W = i2hd_w; Kact = 65;  Nt = 36; base = OFS_I2HD; }
  else if (idx < OFS_H2OD) { W = h2hd_w; Kact = 576; Nt = 36; base = OFS_H2HD; }
  else                     { W = h2od_w; Kact = 576; Nt = 4;  base = OFS_H2OD; }
  size_t local = idx - base;
  int tile = (int)(local >> 9);           // 512 elements per 32x16 tile
  int lane = (int)((local >> 4) & 31);
  int j    = (int)(local & 15);
  int kt = tile / Nt, nt = tile % Nt;
  int n  = nt * 16 + (lane & 15);
  int k  = kt * 32 + 8 * (lane >> 4) + (j < 8 ? j : 8 + j);   // interleaved-K B layout (matches A)
  float v = (k < Kact) ? W[(size_t)n * Kact + k] : 0.0f;
  dst[idx] = f2bf(v);
}

// ---------------------------------------------------------------------------
// Persistent RNN kernel: one workgroup per 16 batch rows, all T steps local.
// ---------------------------------------------------------------------------
extern "C" __global__ void __launch_bounds__(NTHREADS, 1)
millies_rnn_kernel(const float* __restrict__ data,
                   const float* __restrict__ h0_v,
                   const float* __restrict__ h0_m,
                   const float* __restrict__ i2h_b,
                   const float* __restrict__ h2h_b,
                   const float* __restrict__ h2o_b,
                   const float* __restrict__ thal_b,
                   const float* __restrict__ i2hd_b,
                   const float* __restrict__ h2hd_b,
                   const float* __restrict__ h2od_b,
                   const unsigned short* __restrict__ wpack,
                   float* __restrict__ out)
{
  const int tid  = threadIdx.x;
  const int lane = tid & 31;
  const int wv   = tid >> 5;       // 0..8
  const int half = lane >> 4;      // 0/1
  const int lm   = lane & 15;      // M for A-frags, N-sub for C/D frags
  const int b0   = blockIdx.x * BT;

  __shared__ alignas(16) unsigned short sHv[BT * HP];     // h_v bf16 (row stride HP)
  __shared__ alignas(16) unsigned short sHm[BT * HP];     // h_m bf16
  __shared__ alignas(16) unsigned short sX[BT * L_];      // x tile bf16 (stride 128)
  __shared__ alignas(16) unsigned short sThal[BT * 160];  // [out_v|hold|prev_m|0] bf16
  __shared__ alignas(16) unsigned short sMot[BT * 96];    // [out_t|hold|0] bf16
  __shared__ float sPrevT[BT * O_];
  __shared__ float sPrevM[BT * O_];
  __shared__ float sOutV[BT * O_];
  __shared__ float sHold[BT];

  // ---- init hidden state + feedback buffers ----
  for (int i = tid; i < BT * H_; i += NTHREADS) {
    int r = i / H_, c = i % H_;
    sHv[r * HP + c] = f2bf(h0_v[(size_t)(b0 + r) * H_ + c]);
    sHm[r * HP + c] = f2bf(h0_m[(size_t)(b0 + r) * H_ + c]);
  }
  for (int i = tid; i < BT * O_; i += NTHREADS) { sPrevT[i] = 0.0f; sPrevM[i] = 0.0f; }

  // ---- hoist per-lane biases (read once, live in VGPRs for the whole chain) ----
  float biasV[4], biasM[4];
#pragma unroll
  for (int j = 0; j < 4; ++j) {
    int n = (wv * 4 + j) * 16 + lm;
    biasV[j] = h2h_b[n] + (n < 512 ? i2h_b[n] : 0.0f);
    biasM[j] = i2hd_b[n] + h2hd_b[n];
  }
  const int n4 = (wv < 4 ? wv * 16 : 0) + lm;    // N for the O=64 heads (waves 0..3)
  const float bias_ov = h2o_b[n4];
  const float bias_ot = thal_b[n4];
  const float bias_om = h2od_b[n4];

  const unsigned short* arowHv = sHv + lm * HP;
  const unsigned short* arowHm = sHm + lm * HP;
  const unsigned short* arowX  = sX + lm * L_;
  const unsigned short* arowTh = sThal + lm * 160;
  const unsigned short* arowMo = sMot + lm * 96;
  const int ntb = wv * 4;                        // first of this wave's 4 N-tiles

  for (int t = 0; t < T_; ++t) {
    __syncthreads();
    // ---- stage x tile (fp32 -> bf16) into LDS + hold column ----
    for (int i = tid; i < BT * L_; i += NTHREADS) {
      int r = i >> 7, cc = i & (L_ - 1);
      sX[i] = f2bf(data[((size_t)(b0 + r) * T_ + t) * L_ + cc]);
    }
    if (tid < BT) sHold[tid] = data[((size_t)(b0 + tid) * T_ + t) * L_ + (L_ - 1)];
    __syncthreads();

    // ===== phase 1: h_v = retanh([x@i2h^T + b | prev_t] + h_v@h2h^T + b) =====
    {
      v8f c[4];
#pragma unroll
      for (int j = 0; j < 4; ++j) {
        const int n = (ntb + j) * 16 + lm;
#pragma unroll
        for (int r = 0; r < 8; ++r) c[j][r] = biasV[j];
        if (n >= 512) {                               // prev_t feedback columns
#pragma unroll
          for (int r = 0; r < 8; ++r) c[j][r] += sPrevT[(r + 8 * half) * O_ + (n - 512)];
        }
      }
      if (wv < 8) {                                   // x @ i2h^T (N<512 only)
#pragma unroll
        for (int kt = 0; kt < 4; ++kt) {
          v16bf a = load_afrag_lds(arowX, kt, half);  // shared across 4 N-tiles
#pragma unroll
          for (int j = 0; j < 4; ++j)
            c[j] = wmma_bf16(a,
                load_bfrag(wpack + OFS_I2H + (size_t)(kt * 32 + ntb + j) * 512, lane), c[j]);
        }
      }
      for (int kt = 0; kt < 18; ++kt) {               // h_v @ h2h^T
        v16bf a = load_afrag_lds(arowHv, kt, half);   // one A frag, 4 WMMAs
#pragma unroll
        for (int j = 0; j < 4; ++j)
          c[j] = wmma_bf16(a,
              load_bfrag(wpack + OFS_H2H + (size_t)(kt * 36 + ntb + j) * 512, lane), c[j]);
      }
      __syncthreads();                                // all reads of sHv done
#pragma unroll
      for (int j = 0; j < 4; ++j) {
        const int n = (ntb + j) * 16 + lm;
#pragma unroll
        for (int r = 0; r < 8; ++r)
          sHv[(r + 8 * half) * HP + n] = f2bf(retanh_f(c[j][r]));
      }
      __syncthreads();
    }

    // ===== phase 2: out_v = h_v@h2o^T + b  (waves 0..3, one N-tile each) =====
    if (wv < 4) {
      v8f c;
#pragma unroll
      for (int r = 0; r < 8; ++r) c[r] = bias_ov;
      for (int kt = 0; kt < 18; ++kt)
        c = wmma_bf16(load_afrag_lds(arowHv, kt, half),
                      load_bfrag(wpack + OFS_H2O + (size_t)(kt * 4 + wv) * 512, lane), c);
#pragma unroll
      for (int r = 0; r < 8; ++r) sOutV[(r + 8 * half) * O_ + n4] = c[r];
    }
    __syncthreads();

    // ===== phase 3a: stage inp_thal = [out_v | hold | prev_m | 0] in bf16 =====
    for (int i = tid; i < BT * 160; i += NTHREADS) {
      int r = i / 160, cc = i % 160;
      float v;
      if      (cc < 64)  v = sOutV[r * O_ + cc];
      else if (cc == 64) v = sHold[r];
      else if (cc < 129) v = sPrevM[r * O_ + (cc - 65)];
      else               v = 0.0f;
      sThal[r * 160 + cc] = f2bf(v);
    }
    for (int i = tid; i < BT * 32; i += NTHREADS) {   // static part of inp_mot
      int r = i / 32, cc = 64 + (i % 32);
      sMot[r * 96 + cc] = (cc == 64) ? f2bf(sHold[r]) : (unsigned short)0;
    }
    __syncthreads();

    // ===== phase 3b: out_t = retanh(inp_thal@thal^T + b)  (waves 0..3) =====
    if (wv < 4) {
      v8f c;
#pragma unroll
      for (int r = 0; r < 8; ++r) c[r] = bias_ot;
#pragma unroll
      for (int kt = 0; kt < 5; ++kt)
        c = wmma_bf16(load_afrag_lds(arowTh, kt, half),
                      load_bfrag(wpack + OFS_THAL + (size_t)(kt * 4 + wv) * 512, lane), c);
#pragma unroll
      for (int r = 0; r < 8; ++r) {
        float v = retanh_f(c[r]);
        int m = r + 8 * half;
        sPrevT[m * O_ + n4] = v;          // feedback into next step's phase 1
        sMot[m * 96 + n4]   = f2bf(v);    // inp_mot columns 0..63
      }
    }
    __syncthreads();

    // ===== phase 4: h_m = retanh(inp_mot@i2hd^T + b + h_m@h2hd^T + b) =====
    {
      v8f c[4];
#pragma unroll
      for (int j = 0; j < 4; ++j) {
#pragma unroll
        for (int r = 0; r < 8; ++r) c[j][r] = biasM[j];
      }
#pragma unroll
      for (int kt = 0; kt < 3; ++kt) {                // inp_mot @ i2hd^T
        v16bf a = load_afrag_lds(arowMo, kt, half);
#pragma unroll
        for (int j = 0; j < 4; ++j)
          c[j] = wmma_bf16(a,
              load_bfrag(wpack + OFS_I2HD + (size_t)(kt * 36 + ntb + j) * 512, lane), c[j]);
      }
      for (int kt = 0; kt < 18; ++kt) {               // h_m @ h2hd^T
        v16bf a = load_afrag_lds(arowHm, kt, half);
#pragma unroll
        for (int j = 0; j < 4; ++j)
          c[j] = wmma_bf16(a,
              load_bfrag(wpack + OFS_H2HD + (size_t)(kt * 36 + ntb + j) * 512, lane), c[j]);
      }
      __syncthreads();                                // all reads of sHm done
#pragma unroll
      for (int j = 0; j < 4; ++j) {
        const int n = (ntb + j) * 16 + lm;
#pragma unroll
        for (int r = 0; r < 8; ++r)
          sHm[(r + 8 * half) * HP + n] = f2bf(retanh_f(c[j][r]));
      }
      __syncthreads();
    }

    // ===== phase 5: out_m = h_m@h2od^T + b -> global + prev_m  (waves 0..3) =====
    if (wv < 4) {
      v8f c;
#pragma unroll
      for (int r = 0; r < 8; ++r) c[r] = bias_om;
      for (int kt = 0; kt < 18; ++kt)
        c = wmma_bf16(load_afrag_lds(arowHm, kt, half),
                      load_bfrag(wpack + OFS_H2OD + (size_t)(kt * 4 + wv) * 512, lane), c);
#pragma unroll
      for (int r = 0; r < 8; ++r) {
        int m = r + 8 * half;
        float v = c[r];
        sPrevM[m * O_ + n4] = v;
        out[((size_t)(b0 + m) * T_ + t) * O_ + n4] = v;
      }
    }
    // loop-top __syncthreads() protects sPrevM/sHold/sX reuse next step
  }
}

extern "C" void kernel_launch(void* const* d_in, const int* in_sizes, int n_in,
                              void* d_out, int out_size, void* d_ws, size_t ws_size,
                              hipStream_t stream) {
  (void)in_sizes; (void)n_in; (void)out_size; (void)ws_size;
  const float* data   = (const float*)d_in[0];
  const float* h0_v   = (const float*)d_in[1];
  const float* h0_m   = (const float*)d_in[2];
  const float* i2h_w  = (const float*)d_in[3];
  const float* i2h_b  = (const float*)d_in[4];
  const float* h2h_w  = (const float*)d_in[5];
  const float* h2h_b  = (const float*)d_in[6];
  const float* h2o_w  = (const float*)d_in[7];
  const float* h2o_b  = (const float*)d_in[8];
  const float* thal_w = (const float*)d_in[9];
  const float* thal_b = (const float*)d_in[10];
  const float* i2hd_w = (const float*)d_in[11];
  const float* i2hd_b = (const float*)d_in[12];
  const float* h2hd_w = (const float*)d_in[13];
  const float* h2hd_b = (const float*)d_in[14];
  const float* h2od_w = (const float*)d_in[15];
  const float* h2od_b = (const float*)d_in[16];
  unsigned short* wpack = (unsigned short*)d_ws;
  float* out = (float*)d_out;

  const int pblocks = (int)((WPACK_TOTAL + 255) / 256);
  pack_weights_kernel<<<pblocks, 256, 0, stream>>>(i2h_w, h2h_w, h2o_w, thal_w,
                                                   i2hd_w, h2hd_w, h2od_w, wpack);
  millies_rnn_kernel<<<B_ / BT, NTHREADS, 0, stream>>>(data, h0_v, h0_m,
      i2h_b, h2h_b, h2o_b, thal_b, i2hd_b, h2hd_b, h2od_b, wpack, out);
}
